// Triplet_Hard_Mining_86887188398250
// MI455X (gfx1250) — compile-verified, matrix operand order checked by reading, not run on hardware
//
#include <hip/hip_runtime.h>
#include <hip/hip_bf16.h>

#define N_SAMP 4096
#define D_FEAT 512
#define MARGIN_F 0.3f

typedef __attribute__((ext_vector_type(16))) __bf16 v16bf;
typedef __attribute__((ext_vector_type(8)))  __bf16 v8bf;
typedef __attribute__((ext_vector_type(8)))  float  v8f;

// ---------------------------------------------------------------------------
// Kernel 1: split fp32 features into bf16 hi/lo planes (x ~= hi + lo exactly
// to ~2^-17 relative). Row-major [N, D] planes.
// ---------------------------------------------------------------------------
__global__ void split_bf16x2_kernel(const float* __restrict__ src,
                                    const float* __restrict__ tgt,
                                    __bf16* __restrict__ sH, __bf16* __restrict__ sL,
                                    __bf16* __restrict__ tH, __bf16* __restrict__ tL) {
    int i = blockIdx.x * blockDim.x + threadIdx.x;
    const int stride = gridDim.x * blockDim.x;
    for (; i < N_SAMP * D_FEAT; i += stride) {
        float x = src[i];
        __bf16 h = (__bf16)x;
        sH[i] = h;
        sL[i] = (__bf16)(x - (float)h);
        float y = tgt[i];
        __bf16 g = (__bf16)y;
        tH[i] = g;
        tL[i] = (__bf16)(y - (float)g);
    }
}

// ---------------------------------------------------------------------------
// Kernel 2: per-row squared norms (one wave32 per row) + re-init ap/an.
// dist >= 0, so fp32 bit patterns order correctly as u32: ap init 0,
// an init +inf.
// ---------------------------------------------------------------------------
__global__ void norms_init_kernel(const float* __restrict__ src,
                                  const float* __restrict__ tgt,
                                  float* __restrict__ e1, float* __restrict__ e2,
                                  unsigned* __restrict__ ap, unsigned* __restrict__ an) {
    const int lane = threadIdx.x & 31;
    const int row  = blockIdx.x * (blockDim.x >> 5) + (threadIdx.x >> 5);
    if (row >= N_SAMP) return;
    float s1 = 0.f, s2 = 0.f;
    for (int k = lane; k < D_FEAT; k += 32) {
        float a = src[row * D_FEAT + k]; s1 += a * a;
        float b = tgt[row * D_FEAT + k]; s2 += b * b;
    }
#pragma unroll
    for (int off = 16; off > 0; off >>= 1) {
        s1 += __shfl_xor(s1, off, 32);
        s2 += __shfl_xor(s2, off, 32);
    }
    if (lane == 0) {
        e1[row] = s1;
        e2[row] = s2;
        ap[row] = 0u;            // max accumulator (dist >= 0 always reachable: diagonal)
        an[row] = 0x7F800000u;   // +inf for min accumulator
    }
}

// ---------------------------------------------------------------------------
// Fragment loader: A row / B column layout for V_WMMA_*_16X16X32_BF16
// (wave32). lane&15 selects the row (A) / column (B); lane>>4 selects the
// K sub-range: elements [k0..k0+7] -> VGPR0-3, [k0+16..k0+23] -> VGPR4-7,
// with k0 = k + (lane>>4)*8. Two contiguous 16-byte loads.
// ---------------------------------------------------------------------------
__device__ inline v16bf load_frag(const __bf16* __restrict__ p) {
    v8bf a = *reinterpret_cast<const v8bf*>(p);
    v8bf b = *reinterpret_cast<const v8bf*>(p + 16);
    v16bf r;
#pragma unroll
    for (int i = 0; i < 8; ++i) { r[i] = a[i]; r[i + 8] = b[i]; }
    return r;
}

// ---------------------------------------------------------------------------
// Kernel 3: GEMM (bf16x3 split-precision) + fused hard-mining epilogue.
// Block = 8 waves arranged 2(M) x 4(N); wave tile = 32x64 -> block 64x256.
// Each wave: 2x4 accumulators of v8f; 24 WMMAs per K=32 step.
// ---------------------------------------------------------------------------
__global__ __launch_bounds__(256)
void dist_minmax_kernel(const __bf16* __restrict__ sH, const __bf16* __restrict__ sL,
                        const __bf16* __restrict__ tH, const __bf16* __restrict__ tL,
                        const float* __restrict__ e1, const float* __restrict__ e2,
                        const int* __restrict__ labels,
                        unsigned* __restrict__ ap, unsigned* __restrict__ an) {
    const int lane   = threadIdx.x & 31;
    const int wave   = threadIdx.x >> 5;
    const int wm     = wave >> 2;          // 0..1
    const int wn     = wave & 3;           // 0..3
    const int row0   = blockIdx.y * 64  + wm * 32;
    const int col0   = blockIdx.x * 256 + wn * 64;
    const int laneLo = lane & 15;
    const int laneHi = lane >> 4;

    const v8f vzero = {0.f, 0.f, 0.f, 0.f, 0.f, 0.f, 0.f, 0.f};
    v8f acc[2][4];
#pragma unroll
    for (int m = 0; m < 2; ++m)
#pragma unroll
        for (int n = 0; n < 4; ++n) acc[m][n] = vzero;

    const int rA0 = row0 + laneLo;
    const int cB0 = col0 + laneLo;

    for (int k = 0; k < D_FEAT; k += 32) {
        const int k0 = k + laneHi * 8;
        v16bf aH[2], aL[2], bH[4], bL[4];
#pragma unroll
        for (int m = 0; m < 2; ++m) {
            const size_t o = (size_t)(rA0 + m * 16) * D_FEAT + k0;
            aH[m] = load_frag(sH + o);
            aL[m] = load_frag(sL + o);
        }
#pragma unroll
        for (int n = 0; n < 4; ++n) {
            const size_t o = (size_t)(cB0 + n * 16) * D_FEAT + k0;
            bH[n] = load_frag(tH + o);
            bL[n] = load_frag(tL + o);
        }
#pragma unroll
        for (int m = 0; m < 2; ++m)
#pragma unroll
            for (int n = 0; n < 4; ++n) {
                acc[m][n] = __builtin_amdgcn_wmma_f32_16x16x32_bf16(
                    false, aH[m], false, bH[n], (short)0, acc[m][n], false, false);
                acc[m][n] = __builtin_amdgcn_wmma_f32_16x16x32_bf16(
                    false, aH[m], false, bL[n], (short)0, acc[m][n], false, false);
                acc[m][n] = __builtin_amdgcn_wmma_f32_16x16x32_bf16(
                    false, aL[m], false, bH[n], (short)0, acc[m][n], false, false);
            }
    }

    // ---- epilogue: dist = e1 + e2 - 2*dot, clamp >= 0, masked max/min ----
    int   clab[4];
    float e2c[4];
#pragma unroll
    for (int n = 0; n < 4; ++n) {
        const int c = cB0 + n * 16;
        clab[n] = labels[c];
        e2c[n]  = e2[c];
    }

    const float INF = __uint_as_float(0x7F800000u);
#pragma unroll
    for (int m = 0; m < 2; ++m) {
#pragma unroll
        for (int v = 0; v < 8; ++v) {
            // C/D layout: VGPR v, lanes 0-15 -> M = v, lanes 16-31 -> M = 8+v
            const int   r   = row0 + m * 16 + laneHi * 8 + v;
            const int   rl  = labels[r];
            const float e1r = e1[r];
            float apv = 0.0f, anv = INF;
#pragma unroll
            for (int n = 0; n < 4; ++n) {
                float d = e1r + e2c[n] - 2.0f * acc[m][n][v];
                d = fmaxf(d, 0.0f);
                if (rl == clab[n]) apv = fmaxf(apv, d);
                else               anv = fminf(anv, d);
            }
            // reduce across the 16 lanes of this half (xor masks stay in-half)
#pragma unroll
            for (int off = 1; off < 16; off <<= 1) {
                apv = fmaxf(apv, __shfl_xor(apv, off, 32));
                anv = fminf(anv, __shfl_xor(anv, off, 32));
            }
            if (laneLo == 0) {
                atomicMax(&ap[r], __float_as_uint(apv));
                atomicMin(&an[r], __float_as_uint(anv));
            }
        }
    }
}

// ---------------------------------------------------------------------------
// Kernel 4: loss = mean(relu(ap - an + margin))
// ---------------------------------------------------------------------------
__global__ void loss_kernel(const unsigned* __restrict__ ap,
                            const unsigned* __restrict__ an,
                            float* __restrict__ out) {
    __shared__ float red[256];
    float s = 0.f;
    for (int i = threadIdx.x; i < N_SAMP; i += 256) {
        float a = __uint_as_float(ap[i]);
        float b = __uint_as_float(an[i]);
        float l = a - b + MARGIN_F;   // b == +inf -> -inf -> relu 0 (matches ref)
        s += (l > 0.f) ? l : 0.f;
    }
    red[threadIdx.x] = s;
    __syncthreads();
    for (int off = 128; off > 0; off >>= 1) {
        if (threadIdx.x < off) red[threadIdx.x] += red[threadIdx.x + off];
        __syncthreads();
    }
    if (threadIdx.x == 0) out[0] = red[0] * (1.0f / (float)N_SAMP);
}

// ---------------------------------------------------------------------------
extern "C" void kernel_launch(void* const* d_in, const int* in_sizes, int n_in,
                              void* d_out, int out_size, void* d_ws, size_t ws_size,
                              hipStream_t stream) {
    const float* src    = (const float*)d_in[0];
    const float* tgt    = (const float*)d_in[1];
    const int*   labels = (const int*)d_in[2];
    float*       out    = (float*)d_out;

    char* ws = (char*)d_ws;
    const size_t PLANE = (size_t)N_SAMP * D_FEAT * sizeof(__bf16); // 4 MB
    __bf16*   sH = (__bf16*)(ws);
    __bf16*   sL = (__bf16*)(ws + PLANE);
    __bf16*   tH = (__bf16*)(ws + 2 * PLANE);
    __bf16*   tL = (__bf16*)(ws + 3 * PLANE);
    float*    e1 = (float*)(ws + 4 * PLANE);
    float*    e2 = (float*)(ws + 4 * PLANE + 16384);
    unsigned* ap = (unsigned*)(ws + 4 * PLANE + 2 * 16384);
    unsigned* an = (unsigned*)(ws + 4 * PLANE + 3 * 16384);

    split_bf16x2_kernel<<<1024, 256, 0, stream>>>(src, tgt, sH, sL, tH, tL);
    norms_init_kernel<<<N_SAMP / 8, 256, 0, stream>>>(src, tgt, e1, e2, ap, an);
    dim3 grid(N_SAMP / 256, N_SAMP / 64);
    dist_minmax_kernel<<<grid, 256, 0, stream>>>(sH, sL, tH, tL, e1, e2, labels, ap, an);
    loss_kernel<<<1, 256, 0, stream>>>(ap, an, out);
}
